// MHSA_70531952935360
// MI455X (gfx1250) — matrix operand order repaired and checked by reading, not compiled, not run on hardware
//
#include <hip/hip_runtime.h>
#include <hip/hip_bf16.h>

// ---------------------------------------------------------------------------
// MHSA for MI455X (gfx1250, wave32, WMMA 16x16x32 f16 -> f32 accumulate)
//   B=2, L=2048, D_MODEL=1024, HEADS=16, DK=64
// Pipeline:
//   proj_qkv x3 : Q=[bh][L][64] f16, K=[bh][L][64] f16, V^T=[bh][64][L] f16
//   attn        : flash-attention, async-DMA double-buffered K/V staging
//   proj_out    : O[4096x1024]f16 @ wo -> f32 out + bias
// ---------------------------------------------------------------------------

#define DM    1024
#define LSEQ  2048
#define HEADS 16
#define DKH   64

typedef __attribute__((ext_vector_type(16))) _Float16 v16h;
typedef __attribute__((ext_vector_type(8)))  _Float16 v8h;
typedef __attribute__((ext_vector_type(8)))  float    v8f;

static __device__ __forceinline__ v16h cat8(v8h lo, v8h hi) {
  return __builtin_shufflevector(lo, hi, 0,1,2,3,4,5,6,7,8,9,10,11,12,13,14,15);
}
static __device__ __forceinline__ v8f wmma16(v16h a, v16h b, v8f c) {
  // D = A(16x32 f16) * B(32x16 f16) + C(16x16 f32)
  return __builtin_amdgcn_wmma_f32_16x16x32_f16(false, a, false, b,
                                                (short)0, c, false, false);
}

// --- CDNA5 async global->LDS DMA (ASYNCcnt-tracked), GV addressing mode ----
static __device__ __forceinline__ void async_b128(const void* lds_dst,
                                                  const void* gsrc) {
  const unsigned int       ldsOff = (unsigned int)(uintptr_t)lds_dst; // addr[31:0]
  const unsigned long long gaddr  = (unsigned long long)(uintptr_t)gsrc;
  asm volatile("global_load_async_to_lds_b128 %0, %1, off"
               :: "v"(ldsOff), "v"(gaddr) : "memory");
}
static __device__ __forceinline__ void wait_async(void) {
  asm volatile("s_wait_asynccnt 0x0" ::: "memory");
}
static __device__ __forceinline__ void wait_async_prev(void) {
  // <=8 outstanding: the 8 DMAs of the *current* prefetch may be in flight,
  // everything before (previous block) has completed (in-order done).
  asm volatile("s_wait_asynccnt 0x8" ::: "memory");
}

// ---------------------------------------------------------------------------
// Kernel 1: Y = x @ W + b  (x fp32 -> f16 on the fly, W staged transposed f16)
// Grid: (16 n-blocks == heads, 32 m-blocks), block 256 (8 waves x 16 rows)
// VMODE==0: out[((b*H+h)*L + l)*64 + d]   (Q / K, row-major per head)
// VMODE==1: out[((b*H+h)*64 + d)*L + l]   (V transposed per head)
// ---------------------------------------------------------------------------
#define WT_STRIDE 48   // halves per LDS row (32 k + pad), 96B => 16B aligned

template <int VMODE>
__global__ __launch_bounds__(256)
void proj_qkv_kernel(const float* __restrict__ x, const float* __restrict__ W,
                     const float* __restrict__ bias, _Float16* __restrict__ out)
{
  __shared__ _Float16 Wt[64 * WT_STRIDE];          // 6 KB, [n][k] transposed

  const int tid    = threadIdx.x;
  const int lane   = tid & 31;
  const int wv     = tid >> 5;
  const int lane16 = lane & 15;
  const int laneHi = lane >> 4;
  const int h      = blockIdx.x;                   // head == 64-wide n block
  const int nBase  = h * 64;
  const int mBase  = blockIdx.y * 128 + wv * 16;

  v8f acc[4] = {};

  for (int kb = 0; kb < DM; kb += 32) {
    __syncthreads();
    // --- stage W[kb..kb+31][nBase..nBase+63] -> Wt[n][k] f16 (transposed) ---
    {
      const int flat = tid * 8;                    // 256*8 = 2048 elements
      const int k = flat >> 6;
      const int n = flat & 63;
      const float* wp = W + (size_t)(kb + k) * DM + nBase + n;
#pragma unroll
      for (int e = 0; e < 8; ++e)
        Wt[(n + e) * WT_STRIDE + k] = (_Float16)wp[e];
      if (kb + 32 < DM)                            // L2 prefetch next W tile
        __builtin_prefetch(wp + (size_t)32 * DM, 0, 0);
    }
    __syncthreads();

    // --- A fragment: 16x32 f16 from x (convert f32->f16 in registers) ------
    v16h a;
    {
      const float* xp = x + (size_t)(mBase + lane16) * DM + kb + laneHi * 8;
#pragma unroll
      for (int i = 0; i < 8; ++i) a[i]     = (_Float16)xp[i];       // K=base+0..7
#pragma unroll
      for (int i = 0; i < 8; ++i) a[i + 8] = (_Float16)xp[16 + i];  // K=base+16..23
    }

    // --- batch all 4 B fragments, then 4 WMMAs -----------------------------
    v16h bf[4];
#pragma unroll
    for (int t = 0; t < 4; ++t) {
      const _Float16* bp = &Wt[(t * 16 + lane16) * WT_STRIDE + laneHi * 8];
      bf[t] = cat8(*(const v8h*)bp, *(const v8h*)(bp + 16));
    }
#pragma unroll
    for (int t = 0; t < 4; ++t) acc[t] = wmma16(a, bf[t], acc[t]);
  }

  // --- bias + pack to f16 + store in attention-ready layout ----------------
#pragma unroll
  for (int t = 0; t < 4; ++t) {
    const float bv = bias[nBase + t * 16 + lane16];
    const int d = t * 16 + lane16;
#pragma unroll
    for (int j = 0; j < 8; ++j) {
      const int m = mBase + j + laneHi * 8;        // C layout: M = j + 8*hi
      const int b = m >> 11;                       // / LSEQ
      const int l = m & (LSEQ - 1);
      const float val = acc[t][j] + bv;
      if (VMODE)
        out[(((size_t)b * HEADS + h) * DKH + d) * LSEQ + l] = (_Float16)val;
      else
        out[(((size_t)b * HEADS + h) * LSEQ + l) * DKH + d] = (_Float16)val;
    }
  }
}

// ---------------------------------------------------------------------------
// Kernel 2: streaming (flash) attention, double-buffered async K/V staging.
// Grid: (L/64 query blocks, HEADS, B), block 128 (4 waves x 16 query rows)
// ---------------------------------------------------------------------------
#define KV_STRIDE 72   // 64 + 8 pad halves => 144B rows, 16B aligned, spreads banks
#define NBLK (LSEQ / 64)

__global__ __launch_bounds__(128)
void attn_kernel(const _Float16* __restrict__ Qh, const _Float16* __restrict__ Kh,
                 const _Float16* __restrict__ Vt, const int* __restrict__ mask,
                 _Float16* __restrict__ Oh)
{
  __shared__ _Float16 Kt[2][64 * KV_STRIDE];       // keys x dk   (2 x 9 KB)
  __shared__ _Float16 Vs[2][64 * KV_STRIDE];       // dk   x keys (2 x 9 KB)
  __shared__ _Float16 Ps[4 * 16 * KV_STRIDE];      // per-wave P scratch (9 KB)

  const int tid    = threadIdx.x;
  const int lane   = tid & 31;
  const int wv     = tid >> 5;
  const int lane16 = lane & 15;
  const int laneHi = lane >> 4;
  const int h      = blockIdx.y;
  const int bq     = blockIdx.z;
  const int bh     = bq * HEADS + h;
  const int mBase  = blockIdx.x * 64 + wv * 16;

  // per-thread staging slice: 32 halves of K row r and V^T row r
  const int r   = tid >> 1;
  const int seg = (tid & 1) * 32;
  const _Float16* kSrc = Kh + ((size_t)bh * LSEQ + r) * DKH + seg;
  const _Float16* vSrc = Vt + ((size_t)bh * DKH + r) * LSEQ + seg;

  // Q A-fragments for this wave's 16 rows (reused for every key block)
  v16h aq[2];
  {
    const _Float16* qp = Qh + ((size_t)bh * LSEQ + (mBase + lane16)) * DKH;
#pragma unroll
    for (int c = 0; c < 2; ++c) {
      const int base = c * 32 + laneHi * 8;
      aq[c] = cat8(*(const v8h*)(qp + base), *(const v8h*)(qp + base + 16));
    }
  }

  v8f o[4] = {};
  float mrow[8], lrow[8];
#pragma unroll
  for (int j = 0; j < 8; ++j) { mrow[j] = -1e30f; lrow[j] = 0.f; }

  _Float16* Pw = Ps + wv * 16 * KV_STRIDE;

  // --- prologue: async-DMA key block 0 into buffer 0 -----------------------
  {
    _Float16* kd = &Kt[0][r * KV_STRIDE + seg];
    _Float16* vd = &Vs[0][r * KV_STRIDE + seg];
#pragma unroll
    for (int i = 0; i < 4; ++i) async_b128(kd + i * 8, kSrc + i * 8);
#pragma unroll
    for (int i = 0; i < 4; ++i) async_b128(vd + i * 8, vSrc + i * 8);
  }

  for (int ib = 0; ib < NBLK; ++ib) {
    const int kb = ib * 64;

    // --- prefetch next block into other buffer, wait for current -----------
    if (ib + 1 < NBLK) {
      const int nb = (ib + 1) & 1;
      _Float16* kd = &Kt[nb][r * KV_STRIDE + seg];
      _Float16* vd = &Vs[nb][r * KV_STRIDE + seg];
#pragma unroll
      for (int i = 0; i < 4; ++i) async_b128(kd + i * 8, kSrc + (kb + 64) * DKH + i * 8);
#pragma unroll
      for (int i = 0; i < 4; ++i) async_b128(vd + i * 8, vSrc + (kb + 64) + i * 8);
      wait_async_prev();                 // current block resident (in-order)
    } else {
      wait_async();
    }
    __syncthreads();

    const _Float16* KtC = &Kt[ib & 1][0];
    const _Float16* VsC = &Vs[ib & 1][0];

    // --- S = Q K^T : 16 x 64 stripe, f32 accumulators -----------------------
    v8f s[4] = {};
#pragma unroll
    for (int c = 0; c < 2; ++c) {
      v16h bf[4];
#pragma unroll
      for (int t = 0; t < 4; ++t) {
        const _Float16* bp = &KtC[(t * 16 + lane16) * KV_STRIDE + c * 32 + laneHi * 8];
        bf[t] = cat8(*(const v8h*)bp, *(const v8h*)(bp + 16));
      }
#pragma unroll
      for (int t = 0; t < 4; ++t) s[t] = wmma16(aq[c], bf[t], s[t]);
    }

    // --- scale + mask -------------------------------------------------------
#pragma unroll
    for (int t = 0; t < 4; ++t) {
      const int key = kb + t * 16 + lane16;
#pragma unroll
      for (int j = 0; j < 8; ++j) {
        const int row = mBase + j + laneHi * 8;
        const int mk  = mask[((size_t)bq * LSEQ + row) * LSEQ + key];
        const float sv = s[t][j] * 0.125f;         // 1/sqrt(64)
        s[t][j] = mk ? sv : -1e9f;
      }
    }

    // --- online softmax (row stats replicated across 16-lane halves) -------
#pragma unroll
    for (int j = 0; j < 8; ++j) {
      float v = fmaxf(fmaxf(s[0][j], s[1][j]), fmaxf(s[2][j], s[3][j]));
      v = fmaxf(v, __shfl_xor(v, 1, 32));
      v = fmaxf(v, __shfl_xor(v, 2, 32));
      v = fmaxf(v, __shfl_xor(v, 4, 32));
      v = fmaxf(v, __shfl_xor(v, 8, 32));
      const float mn   = fmaxf(mrow[j], v);
      const float corr = __expf(mrow[j] - mn);
      mrow[j] = mn;
      float rs = 0.f;
#pragma unroll
      for (int t = 0; t < 4; ++t) {
        const float p = __expf(s[t][j] - mn);
        s[t][j] = p;
        rs += p;
      }
      rs += __shfl_xor(rs, 1, 32);
      rs += __shfl_xor(rs, 2, 32);
      rs += __shfl_xor(rs, 4, 32);
      rs += __shfl_xor(rs, 8, 32);
      lrow[j] = lrow[j] * corr + rs;
#pragma unroll
      for (int t = 0; t < 4; ++t) o[t][j] *= corr;
    }

    // --- P: C-layout regs -> A-layout via wave-private LDS ------------------
#pragma unroll
    for (int t = 0; t < 4; ++t)
#pragma unroll
      for (int j = 0; j < 8; ++j)
        Pw[(j + laneHi * 8) * KV_STRIDE + t * 16 + lane16] = (_Float16)s[t][j];
    asm volatile("s_wait_dscnt 0x0" ::: "memory");  // wave-local LDS RAW

    // --- O += P V : 8 WMMAs -------------------------------------------------
#pragma unroll
    for (int c = 0; c < 2; ++c) {
      const _Float16* pp = &Pw[lane16 * KV_STRIDE + c * 32 + laneHi * 8];
      v16h ap = cat8(*(const v8h*)pp, *(const v8h*)(pp + 16));
      v16h bf[4];
#pragma unroll
      for (int t = 0; t < 4; ++t) {
        const _Float16* bp = &VsC[(t * 16 + lane16) * KV_STRIDE + c * 32 + laneHi * 8];
        bf[t] = cat8(*(const v8h*)bp, *(const v8h*)(bp + 16));
      }
#pragma unroll
      for (int t = 0; t < 4; ++t) o[t] = wmma16(ap, bf[t], o[t]);
    }
    __syncthreads();   // everyone done reading buffers before next DMA issue
  }

  // --- normalize + store O as [b][l][d_model] f16 (A-fragment ready) --------
  float inv[8];
#pragma unroll
  for (int j = 0; j < 8; ++j) inv[j] = lrow[j] > 0.f ? 1.f / lrow[j] : 0.f;
#pragma unroll
  for (int t = 0; t < 4; ++t) {
    const int cg = h * DKH + t * 16 + lane16;
#pragma unroll
    for (int j = 0; j < 8; ++j) {
      const int row = mBase + j + laneHi * 8;
      Oh[((size_t)bq * LSEQ + row) * DM + cg] = (_Float16)(o[t][j] * inv[j]);
    }
  }
}

// ---------------------------------------------------------------------------
// Kernel 3: out = O @ wo + bo  (O already f16/A-layout-ready; f32 output)
// ---------------------------------------------------------------------------
__global__ __launch_bounds__(256)
void proj_out_kernel(const _Float16* __restrict__ Oh, const float* __restrict__ W,
                     const float* __restrict__ bias, float* __restrict__ out)
{
  __shared__ _Float16 Wt[64 * WT_STRIDE];

  const int tid    = threadIdx.x;
  const int lane   = tid & 31;
  const int wv     = tid >> 5;
  const int lane16 = lane & 15;
  const int laneHi = lane >> 4;
  const int nBase  = blockIdx.x * 64;
  const int mBase  = blockIdx.y * 128 + wv * 16;

  v8f acc[4] = {};

  for (int kb = 0; kb < DM; kb += 32) {
    __syncthreads();
    {
      const int flat = tid * 8;
      const int k = flat >> 6;
      const int n = flat & 63;
      const float* wp = W + (size_t)(kb + k) * DM + nBase + n;
#pragma unroll
      for (int e = 0; e < 8; ++e)
        Wt[(n + e) * WT_STRIDE + k] = (_Float16)wp[e];
      if (kb + 32 < DM)
        __builtin_prefetch(wp + (size_t)32 * DM, 0, 0);
    }
    __syncthreads();

    const _Float16* ap_ptr = Oh + (size_t)(mBase + lane16) * DM + kb + laneHi * 8;
    v16h a = cat8(*(const v8h*)ap_ptr, *(const v8h*)(ap_ptr + 16));

    v16h bf[4];
#pragma unroll
    for (int t = 0; t < 4; ++t) {
      const _Float16* bp = &Wt[(t * 16 + lane16) * WT_STRIDE + laneHi * 8];
      bf[t] = cat8(*(const v8h*)bp, *(const v8h*)(bp + 16));
    }
#pragma unroll
    for (int t = 0; t < 4; ++t) acc[t] = wmma16(a, bf[t], acc[t]);
  }

#pragma unroll
  for (int t = 0; t < 4; ++t) {
    const float bv = bias[nBase + t * 16 + lane16];
#pragma unroll
    for (int j = 0; j < 8; ++j) {
      const int m = mBase + j + laneHi * 8;
      out[(size_t)m * DM + nBase + t * 16 + lane16] = acc[t][j] + bv;
    }
  }
}

// ---------------------------------------------------------------------------
extern "C" void kernel_launch(void* const* d_in, const int* in_sizes, int n_in,
                              void* d_out, int out_size, void* d_ws, size_t ws_size,
                              hipStream_t stream) {
  (void)in_sizes; (void)n_in; (void)out_size; (void)ws_size;
  const float* x    = (const float*)d_in[0];
  const int*   mask = (const int*)d_in[1];
  const float* wq   = (const float*)d_in[2];
  const float* bq   = (const float*)d_in[3];
  const float* wk   = (const float*)d_in[4];
  const float* bk   = (const float*)d_in[5];
  const float* wv   = (const float*)d_in[6];
  const float* bv   = (const float*)d_in[7];
  const float* wo   = (const float*)d_in[8];
  const float* bo   = (const float*)d_in[9];

  const size_t seg = (size_t)2 * HEADS * LSEQ * DKH;   // 4M halves = 8 MB
  _Float16* Qh = (_Float16*)d_ws;
  _Float16* Kh = Qh + seg;
  _Float16* Vt = Kh + seg;
  _Float16* Oh = Vt + seg;                             // total 32 MB scratch

  dim3 gProj(DM / 64, (2 * LSEQ) / 128, 1), bProj(256);
  proj_qkv_kernel<0><<<gProj, bProj, 0, stream>>>(x, wq, bq, Qh);
  proj_qkv_kernel<0><<<gProj, bProj, 0, stream>>>(x, wk, bk, Kh);
  proj_qkv_kernel<1><<<gProj, bProj, 0, stream>>>(x, wv, bv, Vt);

  dim3 gAttn(LSEQ / 64, HEADS, 2), bAttn(128);
  attn_kernel<<<gAttn, bAttn, 0, stream>>>(Qh, Kh, Vt, mask, Oh);

  dim3 gOut(DM / 64, (2 * LSEQ) / 128, 1), bOut(256);
  proj_out_kernel<<<gOut, bOut, 0, stream>>>(Oh, wo, bo, (float*)d_out);
}
